// TransferNet_68264210202717
// MI455X (gfx1250) — matrix operand order repaired and checked by reading, not compiled, not run on hardware
//
#include <hip/hip_runtime.h>
#include <hip/hip_bf16.h>

// ---------------------------------------------------------------------------
// TransferNet forward for MI455X (gfx1250, wave32).
// Dominant cost: W_e2 (410MB) + ent_emb (120MB) weight streaming -> ~23us at
// 23.3 TB/s. Both big GEMMs run on v_wmma_f32_16x16x32_bf16 (f32->bf16 on the
// fly, f32 accumulation) so they stay HBM-bound instead of VALU-bound.
// GEMM2 streams W_e2 tiles with the Tensor Data Mover (tensor_load_to_lds,
// TENSORcnt-tracked, double-buffered) so the DMA of chunk k+1 overlaps the
// WMMA work on chunk k.
// Sparse 'follow' uses entity-major [N_ENT][64] layout: every triple touches
// one fully-coalesced 256B row; scatter via global_atomic_add_f32.
// ---------------------------------------------------------------------------

#define BSZ   64
#define MAXQ  32
#define DHH   300
#define N_ENT 100000
#define N_REL 18
#define N_TRI 600000
#define D_CLS 1024

typedef __attribute__((ext_vector_type(16))) __bf16        bf16x16;
typedef __attribute__((ext_vector_type(8)))  float         floatx8;
typedef __attribute__((ext_vector_type(4)))  unsigned int  uint4v;
typedef __attribute__((ext_vector_type(8)))  unsigned int  uint8v;

union ABfrag {
    bf16x16      v;
    unsigned int d[8];
};

__device__ __forceinline__ unsigned short f2bf(float f) {
    unsigned int u = __float_as_uint(f);
    unsigned int r = (u + 0x7FFFu + ((u >> 16) & 1u)) >> 16;   // RNE
    return (unsigned short)r;
}

// pack two f32 into one dword of bf16 pairs (lo in [15:0], hi in [31:16])
__device__ __forceinline__ unsigned int pack_bf16(float lo, float hi) {
    unsigned int ul = __float_as_uint(lo), uh = __float_as_uint(hi);
    unsigned int rl = (ul + 0x7FFFu + ((ul >> 16) & 1u)) >> 16;
    unsigned int rh = (uh + 0x7FFFu + ((uh >> 16) & 1u)) & 0xFFFF0000u;
    return rl | rh;
}

// A-matrix 16x32 bf16 layout (ISA 7.12.2): lanes 0-15 row m hold K={0..7,16..23},
// lanes 16-31 row m hold K={8..15,24..31}; dword j holds pair (k, k+1).
__device__ __forceinline__ int a_kbase(int half, int j) {
    int base = (j < 4) ? (2 * j) : (8 + 2 * j);   // j>=4: 16+2(j-4)
    return base + half * 8;
}

// --------------------- Tensor Data Mover (2D f32 tile) ---------------------
// Builds a D# (ISA 8.3/8.4) for a 2D load: tile_dim0=128 cols x tile_dim1=32
// rows of 4-byte data, row stride = N_ENT elements. dim0_rem/dim1_rem are the
// tensor extents measured from the tile start: TDM zero-fills OOB reads, which
// handles the ragged last column block for free.
__device__ __forceinline__ void tdm_load_tile_f32(const float* gptr, unsigned lds_off,
                                                  unsigned dim0_rem, unsigned dim1_rem) {
    unsigned long long ga = (unsigned long long)(size_t)gptr;
    uint4v g0;
    g0.x = 1u;                                            // count=1, user mode
    g0.y = lds_off;                                       // LDS byte address
    g0.z = (unsigned)ga;                                  // global_addr[31:0]
    g0.w = ((unsigned)(ga >> 32) & 0x01FFFFFFu) | (2u << 30);  // ga[56:32] | type=2
    uint8v g1;
    g1[0] = 2u << 16;                                     // data_size=4B, no flags
    g1[1] = (dim0_rem & 0xFFFFu) << 16;                   // tensor_dim0[15:0]
    g1[2] = (dim0_rem >> 16) | ((dim1_rem & 0xFFFFu) << 16);   // dim0 hi | dim1 lo
    g1[3] = (dim1_rem >> 16) | (128u << 16);              // dim1 hi | tile_dim0=128
    g1[4] = 32u;                                          // tile_dim1=32, tile_dim2=0
    g1[5] = (unsigned)N_ENT;                              // tensor_dim0_stride lo
    g1[6] = 0u;                                           // stride0 hi | stride1 lo
    g1[7] = 0u;                                           // stride1 hi
    asm volatile("tensor_load_to_lds %0, %1, null, null"
                 :: "s"(g0), "s"(g1) : "memory");
}

// --------------------------- utility kernels -------------------------------

__global__ void zero_f32(float* __restrict__ p, long long n) {
    long long stride = (long long)gridDim.x * blockDim.x;
    for (long long i = (long long)blockIdx.x * blockDim.x + threadIdx.x; i < n; i += stride)
        p[i] = 0.0f;
}

// e_s [64][N_ENT] -> eT [N_ENT][64], LDS-tiled so both sides are coalesced.
__global__ void transpose_es(const float* __restrict__ es, float* __restrict__ eT) {
    __shared__ float tile[64][65];
    int ent0 = blockIdx.x * 64;
    int tid  = threadIdx.x;
    int e  = tid & 63, bl = tid >> 6;
#pragma unroll
    for (int pass = 0; pass < 16; ++pass) {
        int b = pass * 4 + bl;
        tile[b][e] = (ent0 + e < N_ENT) ? es[(size_t)b * N_ENT + ent0 + e] : 0.0f;
    }
    __syncthreads();
    int b2 = tid & 63, el = tid >> 6;
#pragma unroll
    for (int pass = 0; pass < 16; ++pass) {
        int e2 = pass * 4 + el;
        if (ent0 + e2 < N_ENT) eT[(size_t)(ent0 + e2) * 64 + b2] = tile[b2][e2];
    }
}

// --------------------- per-step controller (tiny dense) --------------------
__global__ __launch_bounds__(320) void step_front(
    const float* __restrict__ qwh,  const float* __restrict__ qemb,
    const float* __restrict__ Wstep, const float* __restrict__ bstep,
    const float* __restrict__ Wcq,   const float* __restrict__ bcq,
    const float* __restrict__ Wca,   const float* __restrict__ bca,
    const float* __restrict__ Wrel,  const float* __restrict__ brel,
    float* __restrict__ lastc, float* __restrict__ relT, int t)
{
    int b = blockIdx.x, tid = threadIdx.x;
    __shared__ float qe[DHH], lc[DHH], qr[DHH], cqv[DHH];
    __shared__ float att[MAXQ];
    __shared__ float rl[N_REL];
    __shared__ float invsum;

    if (tid < DHH) { qe[tid] = qemb[b * DHH + tid]; lc[tid] = lastc[b * DHH + tid]; }
    __syncthreads();

    const float* Ws = Wstep + (size_t)t * DHH * DHH;
    const float* bs = bstep + t * DHH;
    if (tid < DHH) {
        float s = bs[tid];
        for (int d = 0; d < DHH; ++d) s += qe[d] * Ws[d * DHH + tid];
        qr[tid] = tanhf(s);
    }
    __syncthreads();

    if (tid < DHH) {
        float s = bcq[tid];
        for (int d = 0; d < DHH; ++d) s += lc[d] * Wcq[d * DHH + tid];
        for (int d = 0; d < DHH; ++d) s += qr[d] * Wcq[(DHH + d) * DHH + tid];
        cqv[tid] = s;
    }
    __syncthreads();

    if (tid < MAXQ) {
        const float* row = qwh + ((size_t)b * MAXQ + tid) * DHH;
        float s = 0.0f;
        for (int d = 0; d < DHH; ++d) s += cqv[d] * row[d] * Wca[d];
        att[tid] = s + bca[0];
    }
    __syncthreads();
    if (tid == 0) {
        float mx = att[0];
        for (int q = 1; q < MAXQ; ++q) mx = fmaxf(mx, att[q]);
        float sm = 0.0f;
        for (int q = 0; q < MAXQ; ++q) { att[q] = __expf(att[q] - mx); sm += att[q]; }
        float inv = 1.0f / sm;
        for (int q = 0; q < MAXQ; ++q) att[q] *= inv;
    }
    __syncthreads();

    if (tid < DHH) {
        float s = 0.0f;
        for (int q = 0; q < MAXQ; ++q) s += att[q] * qwh[((size_t)b * MAXQ + q) * DHH + tid];
        lc[tid] = s;
        lastc[b * DHH + tid] = s;
    }
    __syncthreads();

    if (tid < N_REL) {
        float s = brel[tid];
        for (int d = 0; d < DHH; ++d) s += lc[d] * Wrel[d * N_REL + tid];
        rl[tid] = s;
    }
    __syncthreads();
    if (tid == 0) {
        float mx = rl[0];
        for (int r = 1; r < N_REL; ++r) mx = fmaxf(mx, rl[r]);
        float sm = 0.0f;
        for (int r = 0; r < N_REL; ++r) { rl[r] = __expf(rl[r] - mx); sm += rl[r]; }
        invsum = 1.0f / sm;
    }
    __syncthreads();
    if (tid < N_REL) relT[tid * 64 + b] = rl[tid] * invsum;
}

// ------------------------------ sparse follow ------------------------------
__global__ void follow_kernel(const float* __restrict__ eIn, const float* __restrict__ relT,
                              const int* __restrict__ subj, const int* __restrict__ rel,
                              const int* __restrict__ obj,  float* __restrict__ eOut,
                              long long total)
{
    long long stride = (long long)gridDim.x * blockDim.x;
    for (long long i = (long long)blockIdx.x * blockDim.x + threadIdx.x; i < total; i += stride) {
        int t = (int)(i >> 6), b = (int)(i & 63);
        int s = subj[t], r = rel[t], o = obj[t];
        float v = eIn[(size_t)s * 64 + b] * relT[r * 64 + b];
        atomicAdd(&eOut[(size_t)o * 64 + b], v);
    }
}

// row sums over entities (for normalization)
__global__ void rowsum_kernel(const float* __restrict__ eT, float* __restrict__ sums) {
    __shared__ float lds[256];
    int tid = threadIdx.x, b = tid & 63, g = tid >> 6;
    float acc = 0.0f;
    for (int ent = blockIdx.x * 4 + g; ent < N_ENT; ent += gridDim.x * 4)
        acc += eT[(size_t)ent * 64 + b];
    lds[tid] = acc;
    __syncthreads();
    if (tid < 64)
        atomicAdd(&sums[tid], lds[tid] + lds[tid + 64] + lds[tid + 128] + lds[tid + 192]);
}

// ------------- pred1: p[64][300] = eT^T @ ent_emb  (WMMA, K-split) ----------
__global__ __launch_bounds__(256) void pred1_wmma(const float* __restrict__ eT,
                                                  const float* __restrict__ E,
                                                  float* __restrict__ p)
{
    __shared__ unsigned short As[64 * 34];    // [m][k] pairs contiguous in k
    __shared__ unsigned short Bs[304 * 34];   // [n][k] pairs contiguous in k
    int tid  = threadIdx.x;
    int wave = tid >> 5, lane = tid & 31;
    int h = lane >> 4, lr = lane & 15;

    const int per = (N_ENT + 63) / 64;        // 1563
    int e0 = blockIdx.x * per;
    int e1 = min(e0 + per, N_ENT);

    floatx8 acc[4][3];
#pragma unroll
    for (int mt = 0; mt < 4; ++mt)
#pragma unroll
        for (int i = 0; i < 3; ++i)
#pragma unroll
            for (int r = 0; r < 8; ++r) acc[mt][i][r] = 0.0f;

    int nts[3] = { wave, wave + 8, wave + 16 };   // 19 n-tiles total

    for (int k0 = e0; k0 < e1; k0 += 32) {
        for (int idx = tid; idx < 64 * 32; idx += 256) {
            int kk = idx >> 6, m = idx & 63;
            int ent = k0 + kk;
            float v = (ent < e1) ? eT[(size_t)ent * 64 + m] : 0.0f;
            As[m * 34 + kk] = f2bf(v);
        }
        for (int idx = tid; idx < 304 * 32; idx += 256) {
            int kk = idx / 304, n = idx - kk * 304;
            int ent = k0 + kk;
            float v = (ent < e1 && n < DHH) ? E[(size_t)ent * DHH + n] : 0.0f;
            Bs[n * 34 + kk] = f2bf(v);
        }
        __syncthreads();

        ABfrag a[4];
#pragma unroll
        for (int mt = 0; mt < 4; ++mt) {
            int m = mt * 16 + lr;
#pragma unroll
            for (int j = 0; j < 8; ++j)
                a[mt].d[j] = *(const unsigned int*)&As[m * 34 + a_kbase(h, j)];
        }
#pragma unroll
        for (int i = 0; i < 3; ++i) {
            int nt = nts[i];
            if (nt >= 19) continue;               // wave-uniform
            ABfrag bf;
            int nloc = nt * 16 + lr;
#pragma unroll
            for (int j = 0; j < 8; ++j)
                bf.d[j] = *(const unsigned int*)&Bs[nloc * 34 + a_kbase(h, j)];
#pragma unroll
            for (int mt = 0; mt < 4; ++mt)
                acc[mt][i] = __builtin_amdgcn_wmma_f32_16x16x32_bf16(
                    false, a[mt].v, false, bf.v, (short)0, acc[mt][i], false, false);
        }
        __syncthreads();
    }

    for (int i = 0; i < 3; ++i) {
        int nt = nts[i];
        if (nt >= 19) continue;
        int n = nt * 16 + lr;
        if (n >= DHH) continue;
#pragma unroll
        for (int mt = 0; mt < 4; ++mt)
#pragma unroll
            for (int r = 0; r < 8; ++r) {
                int m = mt * 16 + r + 8 * h;      // C/D layout: M = r + 8*half
                atomicAdd(&p[m * DHH + n], acc[mt][i][r]);
            }
    }
}

// ------ H = relu((p / (sum+1e-6)) @ W_e1 + b_e1), stored bf16 [64][1024] ----
__global__ void pred2a(const float* __restrict__ p, const float* __restrict__ sums,
                       const float* __restrict__ We1, const float* __restrict__ be1,
                       unsigned short* __restrict__ Hb)
{
    __shared__ float prow[DHH];
    int b = blockIdx.x;
    int c = blockIdx.y * 256 + threadIdx.x;
    for (int i = threadIdx.x; i < DHH; i += 256) prow[i] = p[b * DHH + i];
    __syncthreads();
    float inv = 1.0f / (sums[b] + 1e-6f);
    float s = 0.0f;
    for (int d = 0; d < DHH; ++d) s += prow[d] * We1[(size_t)d * D_CLS + c];
    float hval = fmaxf(s * inv + be1[c], 0.0f);
    Hb[(size_t)b * D_CLS + c] = f2bf(hval);
}

// --------- GEMM2: out[64][100000] = H @ W_e2 + b_e2 (WMMA + TDM) -----------
// 782 blocks x 128 columns. Per k-chunk the 32x128 f32 W_e2 tile is DMA'd by
// the Tensor Data Mover into LDS (double-buffered: TDM of chunk k+1 overlaps
// WMMA of chunk k; TENSORcnt enforces the pipeline). B fragments convert
// f32->bf16 during the LDS->VGPR fragment build. A fragments come straight
// from the L2-hot bf16 H buffer.
__global__ __launch_bounds__(256) void gemm2_wmma(const unsigned short* __restrict__ Hb,
                                                  const float* __restrict__ We2,
                                                  const float* __restrict__ be2,
                                                  float* __restrict__ out)
{
    __shared__ float BsF[2][32][128];          // [buf][k][n] f32 tiles (32KB)
    int tid  = threadIdx.x;
    int wave = tid >> 5, lane = tid & 31;
    int h = lane >> 4, lr = lane & 15;
    long long n0 = (long long)blockIdx.x * 128;
    unsigned dim0_rem = (unsigned)((long long)N_ENT - n0);   // cols from tile start
    unsigned lds_off[2] = { (unsigned)(size_t)&BsF[0][0][0],
                            (unsigned)(size_t)&BsF[1][0][0] };

    floatx8 acc[4];
#pragma unroll
    for (int mt = 0; mt < 4; ++mt)
#pragma unroll
        for (int r = 0; r < 8; ++r) acc[mt][r] = 0.0f;

    if (wave == 0)                              // prologue: DMA chunk 0
        tdm_load_tile_f32(We2 + n0, lds_off[0], dim0_rem, D_CLS);

    for (int kc = 0; kc < 32; ++kc) {
        if (kc < 31) {                          // pipeline: DMA chunk kc+1
            if (wave == 0)
                tdm_load_tile_f32(We2 + (size_t)(kc + 1) * 32 * N_ENT + n0,
                                  lds_off[(kc + 1) & 1], dim0_rem,
                                  (unsigned)(D_CLS - (kc + 1) * 32));
            __builtin_amdgcn_s_wait_tensorcnt(1);   // chunk kc landed
        } else {
            __builtin_amdgcn_s_wait_tensorcnt(0);   // drain
        }
        __syncthreads();

        int k0 = kc * 32;
        const float* Bp = &BsF[kc & 1][0][0];

        ABfrag a[4], bf;
#pragma unroll
        for (int mt = 0; mt < 4; ++mt) {
            int m = mt * 16 + lr;
#pragma unroll
            for (int j = 0; j < 8; ++j)
                a[mt].d[j] = *(const unsigned int*)(Hb + (size_t)m * D_CLS + k0 + a_kbase(h, j));
        }
        {
            int nloc = wave * 16 + lr;
#pragma unroll
            for (int j = 0; j < 8; ++j) {
                int kk = a_kbase(h, j);
                float f0 = Bp[kk * 128 + nloc];        // ds_load_2addr pair
                float f1 = Bp[(kk + 1) * 128 + nloc];
                bf.d[j] = pack_bf16(f0, f1);
            }
        }
#pragma unroll
        for (int mt = 0; mt < 4; ++mt)
            acc[mt] = __builtin_amdgcn_wmma_f32_16x16x32_bf16(
                false, a[mt].v, false, bf.v, (short)0, acc[mt], false, false);
        __syncthreads();                        // buffer reuse fence
    }

    long long n = n0 + wave * 16 + lr;
    if (n < N_ENT) {
        float bias = be2[n];
#pragma unroll
        for (int mt = 0; mt < 4; ++mt)
#pragma unroll
            for (int r = 0; r < 8; ++r) {
                int m = mt * 16 + r + 8 * h;
                out[(size_t)m * N_ENT + n] = acc[mt][r] + bias;
            }
    }
}

// ------------------------------- launcher ----------------------------------
extern "C" void kernel_launch(void* const* d_in, const int* in_sizes, int n_in,
                              void* d_out, int out_size, void* d_ws, size_t ws_size,
                              hipStream_t stream) {
    (void)in_sizes; (void)n_in; (void)out_size; (void)ws_size;
    const float* qwh   = (const float*)d_in[0];
    const float* qemb  = (const float*)d_in[1];
    const float* es    = (const float*)d_in[2];
    const float* Wstep = (const float*)d_in[3];
    const float* bstep = (const float*)d_in[4];
    const float* Wcq   = (const float*)d_in[5];
    const float* bcq   = (const float*)d_in[6];
    const float* Wca   = (const float*)d_in[7];
    const float* bca   = (const float*)d_in[8];
    const float* Wrel  = (const float*)d_in[9];
    const float* brel  = (const float*)d_in[10];
    const float* Eemb  = (const float*)d_in[11];
    const float* We1   = (const float*)d_in[12];
    const float* be1   = (const float*)d_in[13];
    const float* We2   = (const float*)d_in[14];
    const float* be2   = (const float*)d_in[15];
    const int*   subj  = (const int*)d_in[16];
    const int*   rel   = (const int*)d_in[17];
    const int*   obj   = (const int*)d_in[18];
    float* out = (float*)d_out;

    // workspace carve-out (~51.5 MB)
    float* wsf   = (float*)d_ws;
    float* eA    = wsf;                       // [N_ENT][64]
    float* eB    = eA + (size_t)N_ENT * 64;   // [N_ENT][64]
    float* lastc = eB + (size_t)N_ENT * 64;   // [64][300]
    float* relT  = lastc + 64 * DHH;          // [18][64]
    float* sums  = relT + N_REL * 64;         // [64]
    float* p     = sums + 64;                 // [64][300]
    unsigned short* Hb = (unsigned short*)(p + 64 * DHH);   // [64][1024] bf16

    transpose_es<<<(N_ENT + 63) / 64, 256, 0, stream>>>(es, eA);
    zero_f32<<<64, 256, 0, stream>>>(lastc, 64 * DHH);

    float* cur = eA; float* nxt = eB;
    for (int t = 0; t < 3; ++t) {
        step_front<<<BSZ, 320, 0, stream>>>(qwh, qemb, Wstep, bstep, Wcq, bcq,
                                            Wca, bca, Wrel, brel, lastc, relT, t);
        zero_f32<<<4096, 256, 0, stream>>>(nxt, (long long)N_ENT * 64);
        follow_kernel<<<8192, 256, 0, stream>>>(cur, relT, subj, rel, obj, nxt,
                                                (long long)N_TRI * 64);
        float* tmp = cur; cur = nxt; nxt = tmp;
    }

    zero_f32<<<1, 64, 0, stream>>>(sums, 64);
    zero_f32<<<75, 256, 0, stream>>>(p, 64 * DHH);
    rowsum_kernel<<<512, 256, 0, stream>>>(cur, sums);
    pred1_wmma<<<64, 256, 0, stream>>>(cur, Eemb, p);
    pred2a<<<dim3(BSZ, D_CLS / 256), 256, 0, stream>>>(p, sums, We1, be1, Hb);
    gemm2_wmma<<<(N_ENT + 127) / 128, 256, 0, stream>>>(Hb, We2, be2, out);
}
